// AutoregressiveMultiGNNv1_24137716204023
// MI455X (gfx1250) — matrix operand order repaired and checked by reading, not compiled, not run on hardware
//
#include <hip/hip_runtime.h>
#include <cstdint>
#include <cstddef>

typedef __attribute__((ext_vector_type(16))) _Float16 v16h;
typedef __attribute__((ext_vector_type(8)))  float    v8f;

static __device__ __forceinline__ float sigm(float x){ return 1.f/(1.f+__expf(-x)); }

__device__ __forceinline__ float blockReduceSum(float x, float* red){
  red[threadIdx.x]=x; __syncthreads();
  for(int o=blockDim.x>>1;o>0;o>>=1){ if((int)threadIdx.x<o) red[threadIdx.x]+=red[threadIdx.x+o]; __syncthreads(); }
  float r=red[0]; __syncthreads(); return r;
}

// ---------------- WMMA GEMM:  C[MxN] = A[MxKp](f16) * B(packed f16) + bias ----------------
// B packed layout: [Np/16][Kp/32][32 lanes][16 halves], lane l: col n = ntile*16+(l&15),
//                  k = ktile*32 + (l>>4)*16 + h.   A fragment uses the same k order: k = k0 + (l>>4)*16 + s.
// NT = compile-time number of 16-wide N tiles per block (all blocks in a launch identical),
// so the k-loop is branch-free: NT back-to-back WMMAs on distinct accumulators.
template<int NT>
__global__ __launch_bounds__(256) void k_gemm(
    const _Float16* __restrict__ A, const _Float16* __restrict__ Bp,
    const float* __restrict__ bias, float* __restrict__ C,
    int M, int Kp, int N)
{
  __shared__ v16h Bs[NT][32];
  const int wave = threadIdx.x >> 5;
  const int lane = threadIdx.x & 31;
  const int m0 = blockIdx.x*128 + wave*16;
  const int n0 = blockIdx.y*(NT*16);
  const int KT = Kp >> 5;
  v8f acc[NT];
#pragma unroll
  for(int t=0;t<NT;t++)
#pragma unroll
    for(int g=0;g<8;g++) acc[t][g]=0.f;
  const int arow = lane & 15;
  const int ah   = lane >> 4;
  const int mrow = m0 + arow;
  // clamp OOB rows to row 0: their products land in rows that the store guard drops
  const _Float16* Arow = A + (size_t)(mrow < M ? mrow : 0)*Kp;
  for(int k0=0;k0<Kp;k0+=32){
    { int idx=threadIdx.x;
      if(NT*64 == 256 || idx < NT*64){   // constexpr-true for NT=4: no exec-mask branch
        int t=idx>>6; int r=idx&63;
        const _Float16* srcp = Bp + ((size_t)((n0>>4)+t)*KT + (k0>>5))*512 + (r<<3);
        ((float4*)&Bs[t][r>>1])[r&1] = *(const float4*)srcp;
      } }
    __syncthreads();
    union { v16h v; float4 q[2]; } af;
    af.q[0] = *(const float4*)(Arow + k0 + ah*16);
    af.q[1] = *(const float4*)(Arow + k0 + ah*16 + 8);
    // branch-free prefetch: clamp to row start when past the end (harmless, resident)
    __builtin_prefetch(Arow + (k0+64 < Kp ? k0+64 : 0), 0, 1);
#pragma unroll
    for(int t=0;t<NT;t++)
      acc[t] = __builtin_amdgcn_wmma_f32_16x16x32_f16(false, af.v, false, Bs[t][lane],
                                                      (short)0, acc[t], false, false);
    __syncthreads();
  }
  const int rb = (lane>>4)*8, cc = lane&15;
#pragma unroll
  for(int t=0;t<NT;t++){
    int n = n0 + t*16 + cc;
    if(n >= N) continue;
    float b = bias ? bias[n] : 0.f;
#pragma unroll
    for(int g=0;g<8;g++){
      int m = m0 + rb + g;
      if(m < M) C[(size_t)m*N + n] = acc[t][g] + b;
    }
  }
}

// pack f32 row-major W[KxN] -> WMMA B layout (zero padded)
__global__ void k_pack_w(const float* __restrict__ W, _Float16* __restrict__ Bp,
                         int K,int N,int Kp,int Np){
  int idx = blockIdx.x*blockDim.x + threadIdx.x;
  int KT = Kp>>5;
  if(idx >= Np*Kp) return;
  int h = idx & 15;
  int l = (idx>>4) & 31;
  int rest = idx >> 9;
  int kt = rest % KT;
  int nt = rest / KT;
  int n = nt*16 + (l & 15);
  int k = kt*32 + (l>>4)*16 + h;
  float v = (k<K && n<N) ? W[(size_t)k*N + n] : 0.f;
  Bp[idx] = (_Float16)v;
}

// ---------------- LayerNorm (scalar) + vector RMS normalization ----------------
__global__ __launch_bounds__(128) void k_ln(const float* __restrict__ S,const float* __restrict__ V,
   const float* __restrict__ g,const float* __restrict__ b,
   float* __restrict__ So,float* __restrict__ Vo,int sd,int nv)
{
  __shared__ float red[128];
  int row=blockIdx.x;
  const float* s=S+(size_t)row*sd;
  float a=0;
  for(int i=threadIdx.x;i<sd;i+=128) a+=s[i];
  float mu=blockReduceSum(a,red)/sd;
  a=0;
  for(int i=threadIdx.x;i<sd;i+=128){float d=s[i]-mu;a+=d*d;}
  float var=blockReduceSum(a,red)/sd;
  float inv=rsqrtf(var+1e-5f);
  for(int i=threadIdx.x;i<sd;i+=128) So[(size_t)row*sd+i]=(s[i]-mu)*inv*g[i]+b[i];
  const float* v=V+(size_t)row*nv*3;
  a=0;
  for(int j=threadIdx.x;j<nv;j+=128){
    float n2=v[3*j]*v[3*j]+v[3*j+1]*v[3*j+1]+v[3*j+2]*v[3*j+2];
    a+=fmaxf(n2,1e-8f);
  }
  float sc=rsqrtf(blockReduceSum(a,red)/nv);
  for(int i=threadIdx.x;i<nv*3;i+=128) Vo[(size_t)row*nv*3+i]=v[i]*sc;
}

// ---------------- GVP pre: vh = v@wh ; A = f16[ s | |vh| | 0pad ] ----------------
__global__ __launch_bounds__(64) void k_gvp_pre(const float* __restrict__ S,const float* __restrict__ V,
   const float* __restrict__ Wh,_Float16* __restrict__ A,float* __restrict__ VH,
   int si,int vi,int h,int Kp)
{
  int row=blockIdx.x;
  __shared__ float vl[99], vhl[99];
  for(int i=threadIdx.x;i<vi*3;i+=64) vl[i]=V[(size_t)row*vi*3+i];
  __syncthreads();
  for(int idx=threadIdx.x;idx<h*3;idx+=64){
    int hh=idx/3,c=idx-3*hh;
    float a=0;
    for(int k=0;k<vi;k++) a+=vl[k*3+c]*Wh[k*h+hh];
    vhl[idx]=a;
    VH[(size_t)row*h*3+idx]=a;
  }
  __syncthreads();
  const float* sr=S+(size_t)row*si;
  _Float16* ar=A+(size_t)row*Kp;
  for(int i=threadIdx.x;i<Kp;i+=64){
    float val;
    if(i<si) val=sr[i];
    else if(i<si+h){
      int hh=i-si;
      float n2=vhl[hh*3]*vhl[hh*3]+vhl[hh*3+1]*vhl[hh*3+1]+vhl[hh*3+2]*vhl[hh*3+2];
      val=sqrtf(fmaxf(n2,1e-8f));
    } else val=0.f;
    ar[i]=(_Float16)val;
  }
}

// ---------------- GVP post: vout = (vh@wv)*sigmoid(gate); s = SiLU(s) in place ----------------
__global__ __launch_bounds__(64) void k_gvp_post(float* __restrict__ S, const float* __restrict__ G,
    const float* __restrict__ VH, const float* __restrict__ Wv,
    float* __restrict__ Vout, int so,int h,int vo,int act)
{
  int row=blockIdx.x;
  __shared__ float vhl[99];
  for(int i=threadIdx.x;i<h*3;i+=64) vhl[i]=VH[(size_t)row*h*3+i];
  __syncthreads();
  for(int idx=threadIdx.x; idx<vo*3; idx+=64){
    int o=idx/3,c=idx-3*o;
    float a=0;
    for(int k=0;k<h;k++) a+=vhl[k*3+c]*Wv[k*vo+o];
    a *= sigm(G[(size_t)row*vo+o]);
    Vout[(size_t)row*vo*3+idx]=a;
  }
  if(act){
    float* s=S+(size_t)row*so;
    for(int i=threadIdx.x;i<so;i+=64){ float x=s[i]; s[i]=x*sigm(x); }
  }
}

// ---------------- gathers ----------------
__global__ __launch_bounds__(64) void k_gather_enc(const float* __restrict__ sN,const float* __restrict__ vN,
   const float* __restrict__ es,const float* __restrict__ ev,const int* __restrict__ src,
   const int* __restrict__ dst,int base,int rows,int C,float* __restrict__ Sg,float* __restrict__ Vg)
{
  int r=blockIdx.x; if(r>=rows) return;
  int gr=base+r; int e=gr/C; int c=gr-e*C;
  int j=src[e]*C+c, i=dst[e]*C+c;
  const float* sj=sN+(size_t)j*128; const float* si=sN+(size_t)i*128;
  const float* er=es+((size_t)e*C+c)*32;
  float* so=Sg+(size_t)r*288;
  for(int t=threadIdx.x;t<288;t+=64)
    so[t] = (t<128)? sj[t] : (t<160)? er[t-128] : si[t-160];
  const float* vj=vN+(size_t)j*48; const float* vi=vN+(size_t)i*48;
  const float* vr=ev+((size_t)e*C+c)*3;
  float* vo=Vg+(size_t)r*99;
  for(int t=threadIdx.x;t<99;t+=64)
    vo[t] = (t<48)? vj[t] : (t<51)? vr[t-48] : vi[t-51];
}

__global__ __launch_bounds__(64) void k_gather_dec(const float* __restrict__ sN,const float* __restrict__ vN,
   const float* __restrict__ eS,const float* __restrict__ eV,
   const float* __restrict__ esD,const float* __restrict__ evD,
   const int* __restrict__ src,const int* __restrict__ dst,int base,int rows,
   float* __restrict__ Sg,float* __restrict__ Vg)
{
  int r=blockIdx.x; if(r>=rows) return;
  int e=base+r; int js=src[e], di=dst[e]; bool fwd = js<di;
  const float* Sj=(fwd? sN:eS)+(size_t)js*128;
  const float* Si=(fwd? sN:eS)+(size_t)di*128;
  const float* er=esD+(size_t)e*36;
  float* so=Sg+(size_t)r*292;
  for(int t=threadIdx.x;t<292;t+=64)
    so[t] = (t<128)? Sj[t] : (t<164)? er[t-128] : Si[t-164];
  const float* Vj=(fwd? vN:eV)+(size_t)js*48;
  const float* Vi=(fwd? vN:eV)+(size_t)di*48;
  const float* vr=evD+(size_t)e*3;
  float* vo=Vg+(size_t)r*99;
  for(int t=threadIdx.x;t<99;t+=64)
    vo[t] = (t<48)? Vj[t] : (t<51)? vr[t-48] : Vi[t-51];
}

// segment-mean accumulated as atomic add of msg/cnt into residual state
__global__ __launch_bounds__(64) void k_accum(const float* __restrict__ MS,const float* __restrict__ MV,
   const int* __restrict__ dst,const float* __restrict__ cnt,int base,int rows,int C,
   float* __restrict__ S,float* __restrict__ V)
{
  int r=blockIdx.x; if(r>=rows) return;
  int gr=base+r; int e=gr/C; int c=gr-e*C; int d=dst[e];
  float w=1.f/cnt[d];
  float* so=S+((size_t)d*C+c)*128;
  float* vo=V+((size_t)d*C+c)*48;
  const float* ms=MS+(size_t)r*128;
  const float* mv=MV+(size_t)r*48;
  for(int t=threadIdx.x;t<128;t+=64) atomicAdd(&so[t],ms[t]*w);
  for(int t=threadIdx.x;t<48;t+=64)  atomicAdd(&vo[t],mv[t]*w);
}

// ---------------- conf pooling ----------------
__global__ __launch_bounds__(64) void k_pool_nodes(const float* __restrict__ S,const float* __restrict__ V,
   const float* __restrict__ mask,int C,float* __restrict__ sP,float* __restrict__ vP)
{
  int n=blockIdx.x;
  float nct=0; for(int c=0;c<C;c++) nct+=mask[n*C+c];
  for(int t=threadIdx.x;t<128;t+=64){
    float a=0; for(int c=0;c<C;c++) a+=S[((size_t)n*C+c)*128+t]*mask[n*C+c];
    sP[(size_t)n*128+t]=a/nct;
  }
  for(int t=threadIdx.x;t<48;t+=64){
    float a=0; for(int c=0;c<C;c++) a+=V[((size_t)n*C+c)*48+t]*mask[n*C+c];
    vP[(size_t)n*48+t]=a/nct;
  }
}

__global__ __launch_bounds__(64) void k_pool_edges(const float* __restrict__ es,const float* __restrict__ ev,
  const float* __restrict__ mask,const int* __restrict__ src,const int* __restrict__ dst,
  const int* __restrict__ seq,const float* __restrict__ Ws,int C,
  float* __restrict__ esD,float* __restrict__ evD)
{
  int e=blockIdx.x;
  int sj=src[e];
  float nct=0; for(int c=0;c<C;c++) nct+=mask[sj*C+c];
  for(int t=threadIdx.x;t<36;t+=64){
    if(t<32){
      float a=0; for(int c=0;c<C;c++) a+=es[((size_t)e*C+c)*32+t]*mask[sj*C+c];
      esD[(size_t)e*36+t]=a/nct;
    } else {
      float fwd=(src[e]<dst[e])?1.f:0.f;
      esD[(size_t)e*36+t]=Ws[seq[sj]*4+(t-32)]*fwd;
    }
  }
  for(int t=threadIdx.x;t<3;t+=64){
    float a=0; for(int c=0;c<C;c++) a+=ev[((size_t)e*C+c)*3+t]*mask[sj*C+c];
    evD[(size_t)e*3+t]=a/nct;
  }
}

// ---------------- utilities ----------------
__global__ void k_f16(const float* __restrict__ X,_Float16* __restrict__ Y,size_t n){
  size_t i=(size_t)blockIdx.x*blockDim.x+threadIdx.x; if(i<n) Y[i]=(_Float16)X[i];
}
__global__ void k_zero(float* __restrict__ X,size_t n){
  size_t i=(size_t)blockIdx.x*blockDim.x+threadIdx.x; if(i<n) X[i]=0.f;
}
__global__ void k_copy(float* __restrict__ D,const float* __restrict__ S,size_t n){
  size_t i=(size_t)blockIdx.x*blockDim.x+threadIdx.x; if(i<n) D[i]=S[i];
}
__global__ void k_add(float* __restrict__ D,const float* __restrict__ S,size_t n){
  size_t i=(size_t)blockIdx.x*blockDim.x+threadIdx.x; if(i<n) D[i]+=S[i];
}
__global__ void k_cnt(const int* __restrict__ dst,float* __restrict__ cnt,int E){
  int i=blockIdx.x*blockDim.x+threadIdx.x; if(i<E) atomicAdd(&cnt[dst[i]],1.f);
}
__global__ void k_cntfix(float* __restrict__ cnt,int N){
  int i=blockIdx.x*blockDim.x+threadIdx.x; if(i<N) cnt[i]=fmaxf(cnt[i],1.f);
}

// ---------------- host side ----------------
struct GvpH{ const float*wh,*wsb,*wsw,*wsvb,*wsvw,*wv;
             int si,vi,h,so,vo,Kp,NpS,NpG; _Float16 *pws,*pwsv; };
struct LayerH{ GvpH c0,c1,c2,f0,f1; const float*ln0b,*ln0g,*ln1b,*ln1g; };

static inline int imin(int a,int b){return a<b?a:b;}
static inline int al16(int x){return (x+15)&~15;}
static inline int al32(int x){return (x+31)&~31;}

extern "C" void kernel_launch(void* const* d_in, const int* in_sizes, int n_in,
                              void* d_out, int out_size, void* d_ws, size_t ws_size,
                              hipStream_t stream)
{
  (void)n_in; (void)out_size; (void)ws_size;
  const float* node_s=(const float*)d_in[0];
  const float* node_v=(const float*)d_in[1];
  const float* edge_s=(const float*)d_in[2];
  const float* edge_v=(const float*)d_in[3];
  const int*   eidx  =(const int*)  d_in[4];
  const int*   seq   =(const int*)  d_in[5];
  const float* mask  =(const float*)d_in[6];
  const int C = 3;
  const int E = in_sizes[4]/2;
  const int N = in_sizes[6]/C;
  const int* src = eidx;
  const int* dst = eidx + E;
  const int Mn = N*C, Me = E*C;
  const int CH = 30000;

  // ---- bump allocator over workspace ----
  char* wsp=(char*)d_ws; size_t off=0;
  auto alloc=[&](size_t bytes)->void*{ void* p=wsp+off; off=(off+bytes+255)&~(size_t)255; return p; };

  // persistent state
  float* s   =(float*)alloc(sizeof(float)*(size_t)Mn*128);
  float* v   =(float*)alloc(sizeof(float)*(size_t)Mn*48);
  float* sN  =(float*)alloc(sizeof(float)*(size_t)Mn*128);
  float* vN  =(float*)alloc(sizeof(float)*(size_t)Mn*48);
  float* es  =(float*)alloc(sizeof(float)*(size_t)Me*32);
  float* ev  =(float*)alloc(sizeof(float)*(size_t)Me*3);
  float* cnt =(float*)alloc(sizeof(float)*(size_t)N);
  float* sP  =(float*)alloc(sizeof(float)*(size_t)N*128);
  float* vP  =(float*)alloc(sizeof(float)*(size_t)N*48);
  float* sD  =(float*)alloc(sizeof(float)*(size_t)N*128);
  float* vD  =(float*)alloc(sizeof(float)*(size_t)N*48);
  float* eSN =(float*)alloc(sizeof(float)*(size_t)N*128);
  float* eVN =(float*)alloc(sizeof(float)*(size_t)N*48);
  float* esD =(float*)alloc(sizeof(float)*(size_t)E*36);
  float* evD =(float*)alloc(sizeof(float)*(size_t)E*3);

  // shared scratch (sized for max use across stages)
  size_t nA   = (size_t)Me*64;  if((size_t)CH*352>nA) nA=(size_t)CH*352; if((size_t)Mn*544>nA) nA=(size_t)Mn*544;
  size_t nSg  = (size_t)CH*292; if((size_t)Me*32>nSg) nSg=(size_t)Me*32;
  size_t nVg  = (size_t)CH*99;  if((size_t)Me*3 >nVg) nVg=(size_t)Me*3;
  size_t nVH  = (size_t)CH*99;  if((size_t)Mn*96>nVH) nVH=(size_t)Mn*96;
  size_t nS   = (size_t)CH*128; if((size_t)Mn*512>nS) nS=(size_t)Mn*512;
  size_t nV   = (size_t)CH*48;  if((size_t)Mn*96>nV)  nV=(size_t)Mn*96;
  size_t nS16 = (size_t)Me*32;  if((size_t)Mn*512>nS16) nS16=(size_t)Mn*512; if((size_t)CH*128>nS16) nS16=(size_t)CH*128;
  size_t nG   = (size_t)CH*16;  if((size_t)Mn*32>nG) nG=(size_t)Mn*32; if((size_t)Me>nG) nG=(size_t)Me;
  _Float16* bufA   =(_Float16*)alloc(sizeof(_Float16)*nA);
  float*    bufSg  =(float*)   alloc(sizeof(float)*nSg);
  float*    bufVg  =(float*)   alloc(sizeof(float)*nVg);
  float*    bufVH  =(float*)   alloc(sizeof(float)*nVH);
  float*    bufS   =(float*)   alloc(sizeof(float)*nS);
  float*    bufV   =(float*)   alloc(sizeof(float)*nV);
  _Float16* bufSf16=(_Float16*)alloc(sizeof(_Float16)*nS16);
  float*    bufG   =(float*)   alloc(sizeof(float)*nG);

  // ---- GEMM dispatch: pick compile-time NT so every block is branch-free ----
  auto launch_gemm=[&](const _Float16* A,const _Float16* Bp,const float* bias,float* Cc,
                       int M,int Kp,int Nn,int Np){
    int gx=(M+127)/128;
    if(Np>=64){      dim3 g(gx,Np/64); k_gemm<4><<<g,256,0,stream>>>(A,Bp,bias,Cc,M,Kp,Nn); }
    else if(Np==32){ dim3 g(gx,1);     k_gemm<2><<<g,256,0,stream>>>(A,Bp,bias,Cc,M,Kp,Nn); }
    else {           dim3 g(gx,1);     k_gemm<1><<<g,256,0,stream>>>(A,Bp,bias,Cc,M,Kp,Nn); }
  };

  // ---- parameter cursor: jax pytree flatten = sorted dict keys, lists in order ----
  int cur=7;
  auto f=[&]()->const float*{ return (const float*)d_in[cur++]; };
  auto readGvp=[&](int si,int vi,int so,int vo)->GvpH{
    GvpH g{}; g.si=si; g.vi=vi; g.so=so; g.vo=vo;
    g.h=(vi>vo)?vi:vo;
    g.Kp=al32(si+g.h); g.NpS=al16(so); g.NpG=al16(vo>0?vo:1);
    g.wh=f(); g.wsb=f(); g.wsw=f();                 // 'wh', 'ws'{b,w}
    if(vo>0){ g.wsvb=f(); g.wsvw=f(); g.wv=f(); }   // 'wsv'{b,w}, 'wv'
    g.pws=(_Float16*)alloc(sizeof(_Float16)*(size_t)g.NpS*g.Kp);
    { int tot=g.NpS*g.Kp;
      k_pack_w<<<(tot+255)/256,256,0,stream>>>(g.wsw,g.pws,si+g.h,so,g.Kp,g.NpS); }
    if(vo>0){
      g.pwsv=(_Float16*)alloc(sizeof(_Float16)*(size_t)g.NpG*so);
      int tot=g.NpG*so;
      k_pack_w<<<(tot+255)/256,256,0,stream>>>(g.wsvw,g.pwsv,so,vo,so,g.NpG);
    }
    return g;
  };
  auto readLayer=[&](int esd)->LayerH{
    LayerH L;
    L.c0=readGvp(256+esd,33,128,16);
    L.c1=readGvp(128,16,128,16);
    L.c2=readGvp(128,16,128,16);
    L.f0=readGvp(128,16,512,32);
    L.f1=readGvp(512,32,128,16);
    L.ln0b=f(); L.ln0g=f(); L.ln1b=f(); L.ln1g=f();
    return L;
  };

  // sorted top-level keys: W_out, W_s, We_gvp, We_ln, Wv_gvp, Wv_ln, dec, enc
  GvpH Wout = readGvp(128,16,4,0);
  const float* Ws = f();
  GvpH We   = readGvp(32,1,32,1);
  const float* WeLNb=f(); const float* WeLNg=f();
  GvpH Wv   = readGvp(64,4,128,16);
  const float* WvLNb=f(); const float* WvLNg=f();
  LayerH dec[3] = { readLayer(36), readLayer(36), readLayer(36) };
  LayerH enc[3] = { readLayer(32), readLayer(32), readLayer(32) };

  // ---- generic GVP runner ----
  auto run_gvp=[&](GvpH& G, const float* Sin, const float* Vin,
                   float* Slin, float* Vout, int M, int act){
    k_gvp_pre<<<M,64,0,stream>>>(Sin,Vin,G.wh,bufA,bufVH,G.si,G.vi,G.h,G.Kp);
    launch_gemm(bufA,G.pws,G.wsb,Slin,M,G.Kp,G.so,G.NpS);
    if(G.vo>0){
      size_t n=(size_t)M*G.so;
      k_f16<<<(int)((n+255)/256),256,0,stream>>>(Slin,bufSf16,n);
      launch_gemm(bufSf16,G.pwsv,G.wsvb,bufG,M,G.so,G.vo,G.NpG);
      k_gvp_post<<<M,64,0,stream>>>(Slin,bufG,bufVH,G.wv,Vout,G.so,G.h,G.vo,act);
    }
  };

  // ---- input embeddings ----
  k_ln<<<Mn,128,0,stream>>>(node_s,node_v,WvLNg,WvLNb,sN,vN,64,4);
  run_gvp(Wv,sN,vN,s,v,Mn,0);
  k_ln<<<Me,128,0,stream>>>(edge_s,edge_v,WeLNg,WeLNb,bufSg,bufVg,32,1);
  run_gvp(We,bufSg,bufVg,es,ev,Me,0);

  k_zero<<<(N+255)/256,256,0,stream>>>(cnt,(size_t)N);
  k_cnt<<<(E+255)/256,256,0,stream>>>(dst,cnt,E);
  k_cntfix<<<(N+255)/256,256,0,stream>>>(cnt,N);

  // ---- encoder ----
  for(int li=0; li<3; li++){
    LayerH& L = enc[li];
    k_ln<<<Mn,128,0,stream>>>(s,v,L.ln0g,L.ln0b,sN,vN,128,16);
    for(int base=0; base<Me; base+=CH){
      int rows=imin(CH,Me-base);
      k_gather_enc<<<rows,64,0,stream>>>(sN,vN,es,ev,src,dst,base,rows,C,bufSg,bufVg);
      run_gvp(L.c0,bufSg,bufVg,bufS,bufV,rows,1);
      run_gvp(L.c1,bufS,bufV,bufS,bufV,rows,1);
      run_gvp(L.c2,bufS,bufV,bufS,bufV,rows,0);
      k_accum<<<rows,64,0,stream>>>(bufS,bufV,dst,cnt,base,rows,C,s,v);
    }
    k_ln<<<Mn,128,0,stream>>>(s,v,L.ln1g,L.ln1b,sN,vN,128,16);
    run_gvp(L.f0,sN,vN,bufS,bufV,Mn,1);
    run_gvp(L.f1,bufS,bufV,bufS,bufV,Mn,0);
    k_add<<<(int)(((size_t)Mn*128+255)/256),256,0,stream>>>(s,bufS,(size_t)Mn*128);
    k_add<<<(int)(((size_t)Mn*48+255)/256),256,0,stream>>>(v,bufV,(size_t)Mn*48);
  }

  // ---- conformer pooling ----
  k_pool_nodes<<<N,64,0,stream>>>(s,v,mask,C,sP,vP);
  k_pool_edges<<<E,64,0,stream>>>(es,ev,mask,src,dst,seq,Ws,C,esD,evD);
  k_copy<<<(int)(((size_t)N*128+255)/256),256,0,stream>>>(sD,sP,(size_t)N*128);
  k_copy<<<(int)(((size_t)N*48+255)/256),256,0,stream>>>(vD,vP,(size_t)N*48);

  // ---- decoder ----
  for(int li=0; li<3; li++){
    LayerH& L = dec[li];
    k_ln<<<N,128,0,stream>>>(sD,vD,L.ln0g,L.ln0b,sN,vN,128,16);
    k_ln<<<N,128,0,stream>>>(sP,vP,L.ln0g,L.ln0b,eSN,eVN,128,16);
    for(int base=0; base<E; base+=CH){
      int rows=imin(CH,E-base);
      k_gather_dec<<<rows,64,0,stream>>>(sN,vN,eSN,eVN,esD,evD,src,dst,base,rows,bufSg,bufVg);
      run_gvp(L.c0,bufSg,bufVg,bufS,bufV,rows,1);
      run_gvp(L.c1,bufS,bufV,bufS,bufV,rows,1);
      run_gvp(L.c2,bufS,bufV,bufS,bufV,rows,0);
      k_accum<<<rows,64,0,stream>>>(bufS,bufV,dst,cnt,base,rows,1,sD,vD);
    }
    k_ln<<<N,128,0,stream>>>(sD,vD,L.ln1g,L.ln1b,sN,vN,128,16);
    run_gvp(L.f0,sN,vN,bufS,bufV,N,1);
    run_gvp(L.f1,bufS,bufV,bufS,bufV,N,0);
    k_add<<<(int)(((size_t)N*128+255)/256),256,0,stream>>>(sD,bufS,(size_t)N*128);
    k_add<<<(int)(((size_t)N*48+255)/256),256,0,stream>>>(vD,bufV,(size_t)N*48);
  }

  // ---- output head: logits = concat(s, |v@wh|) @ ws + b ----
  k_gvp_pre<<<N,64,0,stream>>>(sD,vD,Wout.wh,bufA,bufVH,128,16,16,Wout.Kp);
  launch_gemm(bufA,Wout.pws,Wout.wsb,(float*)d_out,N,Wout.Kp,4,16);
}
